// MojoFusedLinearCrossEntropyLoss_89773406421729
// MI455X (gfx1250) — compile-verified, matrix operand order checked by reading, not run on hardware
//
#include <hip/hip_runtime.h>

// ---------------------------------------------------------------------------
// Fused linear cross-entropy: loss = mean_over_valid( lse(x@W^T+b) - logit[t] )
//   W: [V,H] f32, x: [N,H] f32, target: [N] int, bias: [V] f32 -> scalar f32
// bf16 WMMA pipe, f32 accumulate; x split into bf16 hi+lo (2 WMMAs) for
// ~fp32 fidelity.  Register blocking: 2 row-tiles x 2 col-tiles per wave so
// every converted W fragment feeds 4 WMMAs.  Per-lane online logsumexp,
// cross-lane/cross-wave merge once at the end.
// ---------------------------------------------------------------------------

typedef __attribute__((ext_vector_type(16))) __bf16        v16bf;
typedef __attribute__((ext_vector_type(8)))  __bf16        v8bf;
typedef __attribute__((ext_vector_type(8)))  float         v8f;
typedef __attribute__((ext_vector_type(4)))  float         v4f;
typedef __attribute__((ext_vector_type(4)))  unsigned int  v4u;

#define IGNORE_INDEX   (-100)
#define H_DIM          4096
#define V_DIM          32000
#define N_TOK          4096
#define M_ROWS         32                       // token rows per workgroup
#define KB             256                      // K slice staged in LDS
#define A_STRIDE       (KB + 8)                 // bf16 elems; rows step 4 banks
#define WAVES          8
#define COLS_PER_WAVE  32                       // two 16-wide WMMA tiles
#define COLS_PER_WG    (WAVES * COLS_PER_WAVE)  // 256
#define VT_ITERS       (V_DIM / COLS_PER_WG)    // 125 (exact)

__device__ __forceinline__ unsigned short bits_from_bf(__bf16 h) {
  return __builtin_bit_cast(unsigned short, h);
}
__device__ __forceinline__ unsigned int pack_bf(__bf16 a, __bf16 b) {
  return (unsigned int)bits_from_bf(a) | ((unsigned int)bits_from_bf(b) << 16);
}

// B fragment (32Kx16N bf16) for one 16-wide column tile; p already includes
// this lane's col*H + K offsets. Lane n<16: K={0..7,16..23}; lanes 16..31:
// K={8..15,24..31}. Two 32B contiguous runs -> 4x v4f + packed cvt.
__device__ __forceinline__ v16bf load_b_frag(const float* __restrict__ p) {
  v4f f0 = *(const v4f*)(p);
  v4f f1 = *(const v4f*)(p + 4);
  v4f f2 = *(const v4f*)(p + 16);
  v4f f3 = *(const v4f*)(p + 20);
  v16bf b;
#pragma unroll
  for (int j = 0; j < 4; ++j) {
    b[j]      = (__bf16)f0[j];
    b[4 + j]  = (__bf16)f1[j];
    b[8 + j]  = (__bf16)f2[j];
    b[12 + j] = (__bf16)f3[j];
  }
  return b;
}

// A fragment from LDS: two aligned 16B loads, zero unpack VALU.
__device__ __forceinline__ v16bf load_a_frag(const unsigned short* __restrict__ p) {
  v8bf g0 = *(const v8bf*)(p);        // K = base .. base+7
  v8bf g1 = *(const v8bf*)(p + 16);   // K = base+16 .. base+23
  v16bf a;
#pragma unroll
  for (int j = 0; j < 8; ++j) { a[j] = g0[j]; a[8 + j] = g1[j]; }
  return a;
}

#define WMMA_BF16(C, A, B) \
  (C) = __builtin_amdgcn_wmma_f32_16x16x32_bf16(false, (A), false, (B), \
                                                (short)0, (C), false, false)

__global__ void __launch_bounds__(256, 1)
flce_main(const float* __restrict__ Wt,    // [V,H]
          const float* __restrict__ X,     // [N,H]
          const int*   __restrict__ tgt,   // [N]
          const float* __restrict__ bias,  // [V]
          float*       __restrict__ ws) {  // ws[0]=sum nll, ws[1]=valid count
  __shared__ unsigned short Ahi[M_ROWS * A_STRIDE];   // bf16(x), row-major, natural k order
  __shared__ unsigned short Alo[M_ROWS * A_STRIDE];   // bf16(x - hi)
  __shared__ float mergeM[WAVES][M_ROWS];
  __shared__ float mergeS[WAVES][M_ROWS];
  __shared__ float mergeT[WAVES][M_ROWS];

  const int tid     = threadIdx.x;
  const int lane    = tid & 31;
  const int wave    = tid >> 5;
  const int half    = lane >> 4;   // 0: lanes 0-15, 1: lanes 16-31
  const int l16     = lane & 15;
  const int rowbase = blockIdx.x * M_ROWS;

  // Slot r<8  -> row rowbase + r + half*8        (row-tile 0)
  // Slot r>=8 -> row rowbase + 16 + (r-8)+half*8 (row-tile 1)
  float runM[16], runS[16], tl[16];
  int   tgR[16];
#pragma unroll
  for (int r = 0; r < 8; ++r) {
    runM[r] = -3.4e38f;     runS[r] = 0.0f;     tl[r] = 0.0f;
    runM[r + 8] = -3.4e38f; runS[r + 8] = 0.0f; tl[r + 8] = 0.0f;
    tgR[r]     = tgt[rowbase + r + half * 8];
    tgR[r + 8] = tgt[rowbase + 16 + r + half * 8];
  }

  for (int vt = 0; vt < VT_ITERS; ++vt) {
    const int vbase = vt * COLS_PER_WG + wave * COLS_PER_WAVE;
    v8f c00 = {}, c01 = {}, c10 = {}, c11 = {};   // [rowtile][coltile]

    for (int kb = 0; kb < H_DIM; kb += KB) {
      __syncthreads();
      // ---- stage x slice: 32 rows x KB, bf16 hi/lo, vectorized b128 stores
      {
        const int row = tid >> 3;           // 32 elems per thread
        const int col = (tid & 7) * 32;
        const float* src = X + (size_t)(rowbase + row) * H_DIM + kb + col;
        unsigned int hbuf[16], lbuf[16];
#pragma unroll
        for (int i = 0; i < 8; ++i) {
          v4f f = *(const v4f*)(src + i * 4);
          __bf16 h0 = (__bf16)f[0], h1 = (__bf16)f[1];
          __bf16 h2 = (__bf16)f[2], h3 = (__bf16)f[3];
          __bf16 l0 = (__bf16)(f[0] - (float)h0), l1 = (__bf16)(f[1] - (float)h1);
          __bf16 l2 = (__bf16)(f[2] - (float)h2), l3 = (__bf16)(f[3] - (float)h3);
          hbuf[i * 2]     = pack_bf(h0, h1);
          hbuf[i * 2 + 1] = pack_bf(h2, h3);
          lbuf[i * 2]     = pack_bf(l0, l1);
          lbuf[i * 2 + 1] = pack_bf(l2, l3);
        }
        v4u* dhi = (v4u*)(Ahi + row * A_STRIDE + col);   // 16B aligned
        v4u* dlo = (v4u*)(Alo + row * A_STRIDE + col);
#pragma unroll
        for (int i = 0; i < 4; ++i) {
          dhi[i] = *(const v4u*)&hbuf[i * 4];
          dlo[i] = *(const v4u*)&lbuf[i * 4];
        }
      }
      __syncthreads();

      // ---- K loop: 16x16x32 WMMA steps; B shared by both row tiles -------
      for (int ks = 0; ks < KB; ks += 32) {
        const int aoff = l16 * A_STRIDE + ks + half * 8;
        v16bf ah0 = load_a_frag(Ahi + aoff);
        v16bf al0 = load_a_frag(Alo + aoff);
        v16bf ah1 = load_a_frag(Ahi + aoff + 16 * A_STRIDE);
        v16bf al1 = load_a_frag(Alo + aoff + 16 * A_STRIDE);

        const int koff = kb + ks + half * 8;
        const float* w0 = Wt + (size_t)(vbase + l16) * H_DIM + koff;
        const float* w1 = Wt + (size_t)(vbase + 16 + l16) * H_DIM + koff;
        v16bf b0 = load_b_frag(w0);
        v16bf b1 = load_b_frag(w1);

        WMMA_BF16(c00, ah0, b0);  WMMA_BF16(c00, al0, b0);
        WMMA_BF16(c10, ah1, b0);  WMMA_BF16(c10, al1, b0);
        WMMA_BF16(c01, ah0, b1);  WMMA_BF16(c01, al0, b1);
        WMMA_BF16(c11, ah1, b1);  WMMA_BF16(c11, al1, b1);
      }
    }

    // ---- bias, branch-free target grab, per-lane online logsumexp ---------
    const int   col0 = vbase + l16;
    const int   col1 = vbase + 16 + l16;
    const float b0v  = bias[col0];
    const float b1v  = bias[col1];
#pragma unroll
    for (int r = 0; r < 8; ++r) {
      {  // row-tile 0 -> slot r
        const float v0 = c00[r] + b0v;
        const float v1 = c01[r] + b1v;
        tl[r] += (tgR[r] == col0) ? v0 : 0.0f;
        tl[r] += (tgR[r] == col1) ? v1 : 0.0f;
        const float nm = fmaxf(runM[r], fmaxf(v0, v1));
        runS[r] = runS[r] * __expf(runM[r] - nm)
                + __expf(v0 - nm) + __expf(v1 - nm);
        runM[r] = nm;
      }
      {  // row-tile 1 -> slot r+8
        const float v0 = c10[r] + b0v;
        const float v1 = c11[r] + b1v;
        tl[r + 8] += (tgR[r + 8] == col0) ? v0 : 0.0f;
        tl[r + 8] += (tgR[r + 8] == col1) ? v1 : 0.0f;
        const float nm = fmaxf(runM[r + 8], fmaxf(v0, v1));
        runS[r + 8] = runS[r + 8] * __expf(runM[r + 8] - nm)
                    + __expf(v0 - nm) + __expf(v1 - nm);
        runM[r + 8] = nm;
      }
    }
  }

  // ---- cross-lane merge (16 lanes per half = one row), then cross-wave ----
#pragma unroll
  for (int r = 0; r < 16; ++r) {
#pragma unroll
    for (int m = 1; m < 16; m <<= 1) {
      const float oM = __shfl_xor(runM[r], m, 32);
      const float oS = __shfl_xor(runS[r], m, 32);
      const float nm = fmaxf(runM[r], oM);
      runS[r] = runS[r] * __expf(runM[r] - nm) + oS * __expf(oM - nm);
      runM[r] = nm;
      tl[r]  += __shfl_xor(tl[r], m, 32);
    }
  }
  if (l16 == 0) {
#pragma unroll
    for (int r = 0; r < 8; ++r) {
      const int s0 = r + half * 8;
      mergeM[wave][s0]      = runM[r];
      mergeS[wave][s0]      = runS[r];
      mergeT[wave][s0]      = tl[r];
      mergeM[wave][16 + s0] = runM[r + 8];
      mergeS[wave][16 + s0] = runS[r + 8];
      mergeT[wave][16 + s0] = tl[r + 8];
    }
  }
  __syncthreads();

  if (tid < M_ROWS) {
    float gm = -3.4e38f;
    for (int w = 0; w < WAVES; ++w) gm = fmaxf(gm, mergeM[w][tid]);
    float s = 0.0f, t = 0.0f;
    for (int w = 0; w < WAVES; ++w) {
      s += mergeS[w][tid] * __expf(mergeM[w][tid] - gm);
      t += mergeT[w][tid];
    }
    const float lse = gm + __logf(s);
    const bool  ok  = (tgt[rowbase + tid] != IGNORE_INDEX);
    atomicAdd(&ws[0], ok ? (lse - t) : 0.0f);
    atomicAdd(&ws[1], ok ? 1.0f : 0.0f);
  }
}

__global__ void flce_zero(float* ws) {
  if (threadIdx.x < 2) ws[threadIdx.x] = 0.0f;
}

__global__ void flce_final(const float* __restrict__ ws, float* __restrict__ out) {
  out[0] = ws[0] / ws[1];
}

extern "C" void kernel_launch(void* const* d_in, const int* in_sizes, int n_in,
                              void* d_out, int out_size, void* d_ws, size_t ws_size,
                              hipStream_t stream) {
  (void)in_sizes; (void)n_in; (void)out_size; (void)ws_size;
  const float* Wt   = (const float*)d_in[0];   // lin_weight [V,H]
  const float* X    = (const float*)d_in[1];   // _input     [N,H]
  const int*   tgt  = (const int*)d_in[2];     // target     [N]
  const float* bias = (const float*)d_in[3];   // bias       [V]
  float* ws  = (float*)d_ws;
  float* out = (float*)d_out;

  flce_zero<<<1, 32, 0, stream>>>(ws);
  flce_main<<<N_TOK / M_ROWS, 256, 0, stream>>>(Wt, X, tgt, bias, ws);
  flce_final<<<1, 1, 0, stream>>>(ws, out);
}
